// GAT_19241453486700
// MI455X (gfx1250) — matrix operand order repaired and checked by reading, not compiled, not run on hardware
//
#include <hip/hip_runtime.h>
#include <hip/hip_bf16.h>

#define NHEADS 8
#define MT 5   // M-tiles per wave; 100000/16 = 6250 = 5*1250 (exact)

typedef __attribute__((ext_vector_type(16))) _Float16 v16h;
typedef __attribute__((ext_vector_type(8)))  float    v8f;

// =====================================================================
// WMMA GEMM: C[M,Nc] = X[M,K] * W[K,Nc] (f32 in memory, f16 MACs, f32
// accumulate). One wave computes MT vertically-stacked 16x16 tiles of a
// column strip, so the B (W) fragment is loaded once per k-step and fed
// to MT WMMAs. K is a template constant -> k-loop fully unrolls.
// Fragment layouts per CDNA5 ISA 7.12.2 (wave32):
//   A 16x32 f16 : lane(0-15)=row M, elems 0..7 = K kbA..kbA+7,
//                 elems 8..15 = K 16+kbA..16+kbA+7, kbA = 8*(lane>>4)
//   B 32x16 f16 : lane&15 = col N, same K order as A, kbB = 8*(lane>>4)
//   C 16x16 f32 : VGPR v -> row v + 8*(lane>>4), col lane&15
// =====================================================================
template<int K>
__global__ void gat_gemm_wmma(const float* __restrict__ X, const float* __restrict__ W,
                              float* __restrict__ C, int M, int Nc) {
  const int lane   = threadIdx.x & 31;
  const int waveId = blockIdx.x * (blockDim.x >> 5) + (threadIdx.x >> 5);
  const int tilesN  = Nc >> 4;
  const int groupsM = M / (16 * MT);
  const int gm = waveId / tilesN;
  const int tn = waveId - gm * tilesN;
  if (gm >= groupsM) return;           // wave-uniform exit (EXEC stays all-ones)

  const int row = lane & 15;
  const int hi  = lane >> 4;
  const int kbA = hi * 8;
  const int col = tn * 16 + row;
  const int kbB = hi * 8;
  const int tm0 = gm * MT;

  const float* Arow[MT];
#pragma unroll
  for (int mt = 0; mt < MT; ++mt)
    Arow[mt] = X + (size_t)((tm0 + mt) * 16 + row) * K;

  v8f acc[MT];
#pragma unroll
  for (int mt = 0; mt < MT; ++mt) acc[mt] = (v8f){};

#pragma unroll
  for (int k0 = 0; k0 < K; k0 += 32) {
    // ---- shared B fragment (16 strided f32 -> f16) ----
    v16h b;
#pragma unroll
    for (int e = 0; e < 16; ++e)
      b[e] = (_Float16)W[(size_t)(k0 + kbB + (e & 7) + ((e >> 3) << 4)) * Nc + col];

    // ---- MT A fragments (4x b128 each) + MT WMMAs ----
#pragma unroll
    for (int mt = 0; mt < MT; ++mt) {
      const float4* p0 = reinterpret_cast<const float4*>(Arow[mt] + k0 + kbA);
      const float4* p1 = reinterpret_cast<const float4*>(Arow[mt] + k0 + kbA + 16);
      float4 q0 = p0[0], q1 = p0[1], q2 = p1[0], q3 = p1[1];
      v16h a;
      a[0]=(_Float16)q0.x;  a[1]=(_Float16)q0.y;  a[2]=(_Float16)q0.z;  a[3]=(_Float16)q0.w;
      a[4]=(_Float16)q1.x;  a[5]=(_Float16)q1.y;  a[6]=(_Float16)q1.z;  a[7]=(_Float16)q1.w;
      a[8]=(_Float16)q2.x;  a[9]=(_Float16)q2.y;  a[10]=(_Float16)q2.z; a[11]=(_Float16)q2.w;
      a[12]=(_Float16)q3.x; a[13]=(_Float16)q3.y; a[14]=(_Float16)q3.z; a[15]=(_Float16)q3.w;
      acc[mt] = __builtin_amdgcn_wmma_f32_16x16x32_f16(false, a, false, b,
                                                       (short)0, acc[mt], false, false);
    }
  }

#pragma unroll
  for (int mt = 0; mt < MT; ++mt)
#pragma unroll
    for (int v = 0; v < 8; ++v)
      C[(size_t)((tm0 + mt) * 16 + v + hi * 8) * Nc + col] = acc[mt][v];
}

// ---------- per-(node,head) attention scores ----------
__global__ void gat_scores(const float* __restrict__ H, const float* __restrict__ a_src,
                           const float* __restrict__ a_dst, float* __restrict__ s_src,
                           float* __restrict__ s_dst, int N, int D) {
  int t = blockIdx.x * blockDim.x + threadIdx.x;
  if (t >= N * NHEADS) return;
  int h = t & (NHEADS - 1);
  int n = t >> 3;
  const float* hp = H + (size_t)n * NHEADS * D + h * D;
  float ss = 0.f, sd = 0.f;
  for (int d = 0; d < D; ++d) { float v = hp[d]; ss += v * a_src[h*D+d]; sd += v * a_dst[h*D+d]; }
  s_src[t] = ss; s_dst[t] = sd;
}

// ---------- order-preserving f32 <-> u32 encoding for atomicMax ----------
__device__ __forceinline__ unsigned encf(float x) {
  unsigned u = __float_as_uint(x);
  return (u & 0x80000000u) ? ~u : (u | 0x80000000u);
}
__device__ __forceinline__ float decf(unsigned e) {
  return __uint_as_float((e & 0x80000000u) ? (e & 0x7fffffffu) : ~e);
}
__device__ __forceinline__ float lrelu(float v) { return v > 0.f ? v : 0.2f * v; }

__global__ void fill_u32(unsigned* __restrict__ p, unsigned v, int n) {
  int t = blockIdx.x * blockDim.x + threadIdx.x;
  if (t < n) p[t] = v;
}

__device__ __forceinline__ void edge_ends(const int* __restrict__ ei, int NE, int e,
                                          int& s, int& d) {
  if (e < NE) { s = ei[e]; d = ei[NE + e]; }   // edge_index[0,:]=src, [1,:]=dst
  else        { s = d = e - NE; }              // appended self-loops
}

// ---------- phase 1: segment max over destinations ----------
__global__ void edge_max(const int* __restrict__ ei, int NE, int NN,
                         const float* __restrict__ ssrc, const float* __restrict__ sdst,
                         unsigned* __restrict__ menc) {
  int t = blockIdx.x * blockDim.x + threadIdx.x;
  if (t >= (NE + NN) * NHEADS) return;
  int h = t & (NHEADS - 1), e = t >> 3, s, d;
  edge_ends(ei, NE, e, s, d);
  float v = lrelu(ssrc[s*NHEADS + h] + sdst[d*NHEADS + h]);
  atomicMax(&menc[d*NHEADS + h], encf(v));
}

// ---------- phase 2: softmax denominator ----------
__global__ void edge_sum(const int* __restrict__ ei, int NE, int NN,
                         const float* __restrict__ ssrc, const float* __restrict__ sdst,
                         const unsigned* __restrict__ menc, float* __restrict__ denom) {
  int t = blockIdx.x * blockDim.x + threadIdx.x;
  if (t >= (NE + NN) * NHEADS) return;
  int h = t & (NHEADS - 1), e = t >> 3, s, d;
  edge_ends(ei, NE, e, s, d);
  float v = lrelu(ssrc[s*NHEADS + h] + sdst[d*NHEADS + h]);
  atomicAdd(&denom[d*NHEADS + h], expf(v - decf(menc[d*NHEADS + h])));
}

// ---------- phase 3: weighted scatter-aggregate, one thread per (edge,feature) ----------
template<int F, int LOGF, int LOGD>
__global__ void edge_agg(const int* __restrict__ ei, int NE, int NN,
                         const float* __restrict__ ssrc, const float* __restrict__ sdst,
                         const unsigned* __restrict__ menc, const float* __restrict__ denom,
                         const float* __restrict__ Hlin, float* __restrict__ agg) {
  int t = blockIdx.x * blockDim.x + threadIdx.x;
  if (t >= ((NE + NN) << LOGF)) return;
  int f = t & (F - 1), e = t >> LOGF, s, d;
  edge_ends(ei, NE, e, s, d);
  int h = f >> LOGD;
  float v = lrelu(ssrc[s*NHEADS + h] + sdst[d*NHEADS + h]);
  float alpha = expf(v - decf(menc[d*NHEADS + h])) / (denom[d*NHEADS + h] + 1e-16f);
  atomicAdd(&agg[(size_t)d * F + f], Hlin[(size_t)s * F + f] * alpha);
}

// ---------- bias (+ optional relu) ----------
__global__ void bias_act(const float* __restrict__ in, const float* __restrict__ b,
                         float* __restrict__ out, int n, int F, int relu) {
  int t = blockIdx.x * blockDim.x + threadIdx.x;
  if (t >= n) return;
  float v = in[t] + b[t & (F - 1)];
  out[t] = relu ? fmaxf(v, 0.f) : v;
}

extern "C" void kernel_launch(void* const* d_in, const int* in_sizes, int n_in,
                              void* d_out, int out_size, void* d_ws, size_t ws_size,
                              hipStream_t stream) {
  (void)n_in; (void)out_size; (void)ws_size;
  const float* x   = (const float*)d_in[0];
  const int*   ei  = (const int*)d_in[1];
  const float* W1  = (const float*)d_in[2];
  const float* a1s = (const float*)d_in[3];
  const float* a1d = (const float*)d_in[4];
  const float* b1  = (const float*)d_in[5];
  const float* W2  = (const float*)d_in[6];
  const float* a2s = (const float*)d_in[7];
  const float* a2d = (const float*)d_in[8];
  const float* b2  = (const float*)d_in[9];

  const int N  = in_sizes[0] / 128;   // 100000 nodes
  const int NE = in_sizes[1] / 2;     // 1600000 edges
  const int F1 = 128, F2 = 64;
  const int ET = NE + N;              // edges + self loops
  const unsigned ENC_NEG_INF = 0x007fffffu;   // encf(-INFINITY)

  // workspace (floats): bufA[N*128] | bufB[N*128] | ssrc[N*8] | sdst[N*8] | menc[N*8] | denom[N*8]
  float*    bufA  = (float*)d_ws;
  float*    bufB  = bufA + (size_t)N * 128;
  float*    ssrc  = bufB + (size_t)N * 128;
  float*    sdst  = ssrc + (size_t)N * 8;
  unsigned* menc  = (unsigned*)(sdst + (size_t)N * 8);
  float*    denom = (float*)(menc + (size_t)N * 8);

  auto cdiv = [](long long a, long long b) { return (unsigned)((a + b - 1) / b); };
  const int B = 256;

  // ================= layer 1 =================
  {
    long long waves = (long long)(N / (16 * MT)) * (F1 / 16);
    gat_gemm_wmma<128><<<cdiv(waves, 8), B, 0, stream>>>(x, W1, bufA, N, F1);
  }
  gat_scores<<<cdiv((long long)N*8, B), B, 0, stream>>>(bufA, a1s, a1d, ssrc, sdst, N, 16);
  fill_u32<<<cdiv((long long)N*8, B), B, 0, stream>>>(menc, ENC_NEG_INF, N*8);
  hipMemsetAsync(denom, 0, (size_t)N*8*sizeof(float), stream);
  hipMemsetAsync(bufB, 0, (size_t)N*F1*sizeof(float), stream);
  edge_max<<<cdiv((long long)ET*8, B), B, 0, stream>>>(ei, NE, N, ssrc, sdst, menc);
  edge_sum<<<cdiv((long long)ET*8, B), B, 0, stream>>>(ei, NE, N, ssrc, sdst, menc, denom);
  edge_agg<128,7,4><<<cdiv((long long)ET*F1, B), B, 0, stream>>>(ei, NE, N, ssrc, sdst, menc, denom, bufA, bufB);
  bias_act<<<cdiv((long long)N*F1, B), B, 0, stream>>>(bufB, b1, bufB, N*F1, F1, 1); // relu, in place

  // ================= layer 2 =================
  float* h2   = bufA;                       // N*64 (reuse)
  float* agg2 = bufA + (size_t)N * 64;      // N*64 (reuse, disjoint)
  {
    long long waves = (long long)(N / (16 * MT)) * (F2 / 16);
    gat_gemm_wmma<128><<<cdiv(waves, 8), B, 0, stream>>>(bufB, W2, h2, N, F2);
  }
  gat_scores<<<cdiv((long long)N*8, B), B, 0, stream>>>(h2, a2s, a2d, ssrc, sdst, N, 8);
  fill_u32<<<cdiv((long long)N*8, B), B, 0, stream>>>(menc, ENC_NEG_INF, N*8);
  hipMemsetAsync(denom, 0, (size_t)N*8*sizeof(float), stream);
  hipMemsetAsync(agg2, 0, (size_t)N*F2*sizeof(float), stream);
  edge_max<<<cdiv((long long)ET*8, B), B, 0, stream>>>(ei, NE, N, ssrc, sdst, menc);
  edge_sum<<<cdiv((long long)ET*8, B), B, 0, stream>>>(ei, NE, N, ssrc, sdst, menc, denom);
  edge_agg<64,6,3><<<cdiv((long long)ET*F2, B), B, 0, stream>>>(ei, NE, N, ssrc, sdst, menc, denom, h2, agg2);
  bias_act<<<cdiv((long long)N*F2, B), B, 0, stream>>>(agg2, b2, (float*)d_out, N*F2, F2, 0);
}
